// ContrastiveLoss_11699490914604
// MI455X (gfx1250) — matrix operand order repaired and checked by reading, not compiled, not run on hardware
//
#include <hip/hip_runtime.h>

typedef __attribute__((ext_vector_type(16))) _Float16 v16h;
typedef __attribute__((ext_vector_type(8)))  _Float16 v8h;
typedef __attribute__((ext_vector_type(8)))  float    v8f;
typedef __attribute__((ext_vector_type(4)))  float    v4f;

#define BATCH   4096
#define DIM     512
#define NROWS   8192           // 2*BATCH
#define INV_T   14.285714285714286f   // 1/0.07 ; also the exact row max (diag cos=1)
#define LOG2E   1.4426950408889634f

union V16 { v16h v; v8h p[2]; };

// ---------------------------------------------------------------------------
// Kernel 1: L2-normalize rows of z1|z2 (fp32) -> f16 feature matrix Fh[8192][512]
// One wave per row. Also zero-initializes the scalar output accumulator.
// ---------------------------------------------------------------------------
__global__ __launch_bounds__(256) void nt_xent_normalize(
    const float* __restrict__ z1, const float* __restrict__ z2,
    _Float16* __restrict__ Fh, float* __restrict__ out)
{
    if (blockIdx.x == 0 && threadIdx.x == 0) *out = 0.0f;
    const int wave = threadIdx.x >> 5;
    const int lane = threadIdx.x & 31;
    const int row  = blockIdx.x * 8 + wave;           // grid = 1024 -> rows 0..8191

    const float* src = (row < BATCH) ? (z1 + (size_t)row * DIM)
                                     : (z2 + (size_t)(row - BATCH) * DIM);
    const v4f* s4 = (const v4f*)(src + lane * 16);    // 16 contiguous floats/lane
    v4f a[4];
    float ss = 0.0f;
#pragma unroll
    for (int i = 0; i < 4; ++i) a[i] = s4[i];
#pragma unroll
    for (int i = 0; i < 4; ++i)
#pragma unroll
        for (int j = 0; j < 4; ++j) ss += a[i][j] * a[i][j];
#pragma unroll
    for (int m = 16; m >= 1; m >>= 1) ss += __shfl_xor(ss, m, 32);

    const float scale = 1.0f / fmaxf(sqrtf(ss), 1e-12f);

    v8h h[2];
#pragma unroll
    for (int i = 0; i < 4; ++i)
#pragma unroll
        for (int j = 0; j < 4; ++j)
            h[i >> 1][(i & 1) * 4 + j] = (_Float16)(a[i][j] * scale);

    v8h* d = (v8h*)(Fh + (size_t)row * DIM + lane * 16);
    d[0] = h[0];
    d[1] = h[1];
}

// ---------------------------------------------------------------------------
// Kernel 2: fused F*F^T / T  +  per-row logsumexp  +  positive-pair gather.
// 64 workgroups x 256 threads (8 waves). Wave w owns M-rows [wg*128+w*16, +16).
// A fragments (K=512) preloaded in registers (indices are compile-time
// constants -> must never be dynamically indexed, or they spill to scratch).
// B columns staged into a DOUBLE-BUFFERED LDS tile with CDNA5 async
// global->LDS copies (ASYNCcnt): the next K-half's 8 async b128 loads are
// issued before the current 32 WMMAs, and s_wait_asynccnt 0x8 retires only
// the older batch (async loads complete in order), overlapping L2 fetch
// with the WMMA pipe. The two K-halves are unrolled so `half`/`buf` are
// literals in each section.
// ---------------------------------------------------------------------------
#define LDS_STRIDE 264          // halves per staged row (256 + 8 pad -> conflict-free)
#define NCHUNK     128          // 8192 columns / 64

__global__ __launch_bounds__(256) void nt_xent_fused_loss(
    const _Float16* __restrict__ Fh, float* __restrict__ out)
{
    __shared__ _Float16 Bsh[2][64 * LDS_STRIDE];      // 2 x 33,792 bytes

    const int tid   = threadIdx.x;
    const int wave  = tid >> 5;
    const int lane  = tid & 31;
    const int l15   = lane & 15;
    const int hi    = lane >> 4;                      // 0 or 1 (K-half selector)
    const int mbase = blockIdx.x * 128 + wave * 16;

    // ---- preload this wave's A fragments: 16 rows x K=512 -> 16 x v16h ----
    // A 16x32 f16 layout: lane(l15) = row M; halves 0-7 = K[kh..kh+8),
    // halves 8-15 = K[kh+16..kh+24), kh = 8*hi.
    v16h afrag[16];
    {
        const _Float16* arow = Fh + (size_t)(mbase + l15) * DIM;
        const int kh = hi * 8;
#pragma unroll
        for (int kf = 0; kf < 16; ++kf) {
            V16 u;
            u.p[0] = *(const v8h*)(arow + kf * 32 + kh);
            u.p[1] = *(const v8h*)(arow + kf * 32 + 16 + kh);
            afrag[kf] = u.v;
        }
    }

    // async stage of (chunk c, K-half) -> buffer `buf` (64 rows x 256 halves)
    auto stage = [&](int c, int half, int buf) {
#pragma unroll
        for (int i = 0; i < 8; ++i) {
            const int idx = tid + 256 * i;            // 0..2047
            const int rr  = idx >> 5;                 // staged row 0..63
            const int cc  = (idx & 31) * 8;           // half offset 0..248
            const unsigned lds = (unsigned)(uintptr_t)(&Bsh[buf][rr * LDS_STRIDE + cc]);
            const unsigned long long ga = (unsigned long long)(uintptr_t)
                (Fh + (size_t)(c * 64 + rr) * DIM + half * 256 + cc);
            asm volatile("global_load_async_to_lds_b128 %0, %1, off"
                         :: "v"(lds), "v"(ga) : "memory");
        }
    };

    v8f acc[4];
    // 32 (kf,s) WMMA steps over one staged K-half; half/buf must be literals.
    auto wmma_block = [&](int half, int buf) {
        const _Float16* Bp = &Bsh[buf][0];
        V16 brot[3];
        auto ldb = [&](int step, V16& dst) {
            const int kf = step >> 2, s = step & 3;
            const int koff = kf * 32 + hi * 8;        // B 32x16: lane l15 = col,
                                                      // K [8*hi..+8) and [16+8*hi..+8)
            dst.p[0] = *(const v8h*)(Bp + (s * 16 + l15) * LDS_STRIDE + koff);
            dst.p[1] = *(const v8h*)(Bp + (s * 16 + l15) * LDS_STRIDE + koff + 16);
        };
        ldb(0, brot[0]);
        ldb(1, brot[1]);
#pragma unroll
        for (int step = 0; step < 32; ++step) {
            if (step + 2 < 32) ldb(step + 2, brot[(step + 2) % 3]);
            const int s = step & 3;
            acc[s] = __builtin_amdgcn_wmma_f32_16x16x32_f16(
                false, afrag[half * 8 + (step >> 2)], false, brot[step % 3].v,
                (short)0, acc[s], false, false);
        }
    };

    // running softmax denominators (row max is the known constant INV_T) + positives
    float vsum[8], vpos[8];
#pragma unroll
    for (int r = 0; r < 8; ++r) { vsum[r] = 0.0f; vpos[r] = 0.0f; }

    const int gi0 = mbase + hi * 8;                   // global row for acc index r=0

    stage(0, 0, 0);                                   // prologue fill of buffer 0

    for (int c = 0; c < NCHUNK; ++c) {                // stream all 8192 columns
        const int n0 = c * 64;

        // ---- K-half 0 : consume buf0, prefetch (c, half1) -> buf1 ----
        stage(c, 1, 1);
        asm volatile("s_wait_asynccnt 0x8" ::: "memory");   // retire buf0's batch
        __syncthreads();
        {
            const v8f vzero = {};
#pragma unroll
            for (int s = 0; s < 4; ++s) acc[s] = vzero;
        }
        wmma_block(0, 0);
        __syncthreads();        // buf0 reads retired before it is restaged below

        // ---- K-half 1 : consume buf1, prefetch (c+1, half0) -> buf0 ----
        if (c + 1 < NCHUNK) {
            stage(c + 1, 0, 0);
            asm volatile("s_wait_asynccnt 0x8" ::: "memory");
        } else {
            asm volatile("s_wait_asynccnt 0x0" ::: "memory");
        }
        __syncthreads();
        wmma_block(1, 1);

        // epilogue: C/D layout -> VGPR r, lane l15 = col, hi selects M+8
#pragma unroll
        for (int s = 0; s < 4; ++s) {
            const int gj = n0 + s * 16 + l15;         // global column
#pragma unroll
            for (int r = 0; r < 8; ++r) {
                const float val = acc[s][r] * INV_T;  // logit
                vsum[r] += exp2f((val - INV_T) * LOG2E);
                const int gi = gi0 + r;
                const int pj = (gi < BATCH) ? gi + BATCH : gi - BATCH;
                vpos[r] += (gj == pj) ? val : 0.0f;
            }
        }
        __syncthreads();        // buf1 reads retired before next chunk restages it
    }

    // reduce across the 16-lane group that shares each row
    float total = 0.0f;
#pragma unroll
    for (int r = 0; r < 8; ++r) {
        float s = vsum[r], p = vpos[r];
#pragma unroll
        for (int m = 8; m >= 1; m >>= 1) {
            s += __shfl_xor(s, m, 16);
            p += __shfl_xor(p, m, 16);
        }
        total += (INV_T + __logf(s)) - p;             // LSE_i - s[i, pos(i)]
    }
    if (l15 == 0) atomicAdd(out, total * (1.0f / (float)NROWS));
}

// ---------------------------------------------------------------------------
extern "C" void kernel_launch(void* const* d_in, const int* in_sizes, int n_in,
                              void* d_out, int out_size, void* d_ws, size_t ws_size,
                              hipStream_t stream) {
    (void)in_sizes; (void)n_in; (void)out_size; (void)ws_size;
    const float* z1 = (const float*)d_in[0];
    const float* z2 = (const float*)d_in[1];
    // d_in[2] (labels) is unused by the reference loss.
    float* out = (float*)d_out;
    _Float16* Fh = (_Float16*)d_ws;                   // 8192*512*2 = 8 MB scratch

    nt_xent_normalize<<<NROWS / 8, 256, 0, stream>>>(z1, z2, Fh, out);
    nt_xent_fused_loss<<<NROWS / 128, 256, 0, stream>>>(Fh, out);
}
